// RGCNEmb_17609365914131
// MI455X (gfx1250) — matrix operand order repaired and checked by reading, not compiled, not run on hardware
//
#include <hip/hip_runtime.h>

// ---------------------------------------------------------------------------
// RGCN (basis-decomposed, 2 layers) for MI455X / gfx1250.
// Sizes fixed by the reference:
#define NN   100000
#define RR   32
#define EE   3200000
#define EMB  128
#define HH   32
#define BB   16
#define CC   16
// ---------------------------------------------------------------------------

typedef __bf16 v16bf __attribute__((ext_vector_type(16)));
typedef float  v8f   __attribute__((ext_vector_type(8)));

__device__ __forceinline__ unsigned short f2bf(float f) {
    unsigned int u = __float_as_uint(f);
    unsigned int r = (u >> 16) & 1u;          // round-to-nearest-even
    u += 0x7FFFu + r;
    return (unsigned short)(u >> 16);
}

// ---------------- init / small prep kernels --------------------------------

__global__ void zero_f32(float* __restrict__ p, long long n) {
    long long i = (long long)blockIdx.x * blockDim.x + threadIdx.x;
    if (i < n) p[i] = 0.0f;
}

__global__ void degree_kernel(const int* __restrict__ rel,
                              const int* __restrict__ fr,
                              float* __restrict__ deg) {
    int e = blockIdx.x * blockDim.x + threadIdx.x;
    if (e < EE) {
        long long seg = (long long)rel[e] * NN + fr[e];
        atomicAdd(&deg[seg], 1.0f);
    }
}

__global__ void conv_bf16_kernel(const float* __restrict__ src,
                                 unsigned short* __restrict__ dst, long long n) {
    long long i = (long long)blockIdx.x * blockDim.x + threadIdx.x;
    if (i < n) dst[i] = f2bf(src[i]);
}

// w1T[r][h][e] = sum_b comps1[r,b] * bases1[b,e,h]   (bf16, B-fragment friendly)
__global__ void build_w1t_kernel(const float* __restrict__ comps1,
                                 const float* __restrict__ bases1,
                                 unsigned short* __restrict__ w1T) {
    int r = blockIdx.x;
    for (int idx = threadIdx.x; idx < HH * EMB; idx += blockDim.x) {
        int h = idx / EMB;
        int e = idx % EMB;
        float acc = 0.0f;
        #pragma unroll
        for (int b = 0; b < BB; ++b)
            acc += comps1[r * BB + b] * bases1[((long long)b * EMB + e) * HH + h];
        w1T[((long long)r * HH + h) * EMB + e] = f2bf(acc);
    }
}

// w2[r][h][c] = sum_b comps2[r,b] * bases2[b,h,c]    (f32, 64 KB total)
__global__ void build_w2_kernel(const float* __restrict__ comps2,
                                const float* __restrict__ bases2,
                                float* __restrict__ w2) {
    int r = blockIdx.x;
    for (int idx = threadIdx.x; idx < HH * CC; idx += blockDim.x) {
        int h = idx / CC;
        int c = idx % CC;
        float acc = 0.0f;
        #pragma unroll
        for (int b = 0; b < BB; ++b)
            acc += comps2[r * BB + b] * bases2[((long long)b * HH + h) * CC + c];
        w2[((long long)r * HH + h) * CC + c] = acc;
    }
}

// ---------------- layer-1 GEMM via WMMA bf16 -------------------------------
// xw[r, n, h] = embeddings(bf16)[n, :] . w1T(bf16)[r, h, :]  (f32 accumulate)
//
// Block = 256 threads = 8 waves. Each wave pins ALL B fragments for its
// relation r in VGPRs (2 column tiles x 4 K-blocks = 64 VGPRs; w1T[r] is only
// 8 KB) and sweeps 16-row tiles of the embedding matrix in a grid-stride
// loop. Per tile: all four A K-block fragments are loaded into distinct
// registers first (one 8x b128 clause), so the compiler can stagger
// s_wait_loadcnt and overlap later loads with earlier WMMAs; then 8 WMMAs on
// two independent accumulators (no D->A/B hazard between the pair).

#define GEMM_BLOCKS_X 98   // 98*8 = 784 waves per relation over 6250 row tiles

__global__ __launch_bounds__(256) void gemm_xw_kernel(
        const unsigned short* __restrict__ embB,   // [NN][EMB] bf16
        const unsigned short* __restrict__ w1T,    // [RR][HH][EMB] bf16
        float* __restrict__ xw) {                  // [RR][NN][HH] f32
    const int r    = blockIdx.y;             // relation
    const int wv   = threadIdx.x >> 5;       // wave in block: 0..7
    const int lane = threadIdx.x & 31;
    const int hi   = lane >> 4;              // lane half (ISA table column)
    const int m    = lane & 15;

    union Frag { v16bf v; unsigned int u[8]; };

    // ---- load all B fragments for relation r (kept live across the sweep) --
    Frag b[2][4];
    #pragma unroll
    for (int ct = 0; ct < 2; ++ct) {
        // B 32x16 bf16, this code's k order (as A): lane column N = lane&15, VGPR v holds K pair
        // hi*16 + 2v within the 32-deep K block.
        const unsigned int* brow = reinterpret_cast<const unsigned int*>(
            w1T + ((long long)r * HH + ct * 16 + m) * EMB);
        #pragma unroll
        for (int kb = 0; kb < 4; ++kb)
            #pragma unroll
            for (int v = 0; v < 8; ++v)
                b[ct][kb].u[v] = brow[(kb * 32 + hi * 16 + 2 * v) >> 1];
    }

    const int tiles  = NN / 16;              // 6250
    const int stride = GEMM_BLOCKS_X * 8;    // waves per relation

    for (int tile = blockIdx.x * 8 + wv; tile < tiles; tile += stride) {
        const unsigned int* arow = reinterpret_cast<const unsigned int*>(
            embB + ((long long)tile * 16 + m) * EMB);

        // prefetch next tile's A row into the cache hierarchy
        int ntile = tile + stride;
        if (ntile < tiles)
            __builtin_prefetch(embB + ((long long)ntile * 16 + m) * EMB, 0, 3);

        // ---- issue ALL A loads for this tile first (distinct registers) ----
        Frag a[4];
        #pragma unroll
        for (int kb = 0; kb < 4; ++kb)
            #pragma unroll
            for (int v = 0; v < 8; ++v) {
                // A 16x32 bf16, this code's k order (as B): row M = lane&15, VGPR v holds K pair
                // hi*16 + 2v.
                int ka = kb * 32 + hi * 16 + 2 * v;
                a[kb].u[v] = arow[ka >> 1];
            }

        v8f acc0 = {}, acc1 = {};
        #pragma unroll
        for (int kb = 0; kb < 4; ++kb) {     // K blocks of 32
            acc0 = __builtin_amdgcn_wmma_f32_16x16x32_bf16(
                       false, a[kb].v, false, b[0][kb].v, (short)0, acc0, false, false);
            acc1 = __builtin_amdgcn_wmma_f32_16x16x32_bf16(
                       false, a[kb].v, false, b[1][kb].v, (short)0, acc1, false, false);
        }

        // D 16x16 f32: VGPR j -> row (j + hi*8), column = lane&15 (+16 for acc1)
        float* dst = xw + (((long long)r * NN) + (long long)tile * 16) * HH + m;
        #pragma unroll
        for (int j = 0; j < 8; ++j) {
            long long mrow = j + hi * 8;
            dst[mrow * HH]      = acc0[j];
            dst[mrow * HH + 16] = acc1[j];
        }
    }
}

// ---------------- edge scatter passes --------------------------------------

// hidden1[fr, h] += (1/deg[rel*N+fr]) * xw[rel, to, h]
__global__ void edge1_kernel(const int* __restrict__ rel,
                             const int* __restrict__ fr,
                             const int* __restrict__ to,
                             const float* __restrict__ deg,
                             const float* __restrict__ xw,
                             float* __restrict__ hidden1) {
    long long gid = (long long)blockIdx.x * blockDim.x + threadIdx.x;
    if (gid >= (long long)EE * HH) return;
    int e = (int)(gid >> 5);
    int h = (int)(gid & 31);
    int r = rel[e], f = fr[e], t = to[e];
    float val = 1.0f / deg[(long long)r * NN + f];
    float x   = xw[(((long long)r * NN) + t) * HH + h];
    atomicAdd(&hidden1[(long long)f * HH + h], val * x);
}

__global__ void relu_bias_kernel(float* __restrict__ hidden1,
                                 const float* __restrict__ bias1) {
    long long i = (long long)blockIdx.x * blockDim.x + threadIdx.x;
    if (i < (long long)NN * HH) {
        float v = hidden1[i] + bias1[i & 31];
        hidden1[i] = v > 0.0f ? v : 0.0f;
    }
}

__global__ void out_init_kernel(float* __restrict__ out,
                                const float* __restrict__ bias2) {
    long long i = (long long)blockIdx.x * blockDim.x + threadIdx.x;
    if (i < (long long)NN * CC) out[i] = bias2[i & 15];
}

// out[fr, c] += (1/deg[rel*N+fr]) * sum_h hidden1[to, h] * w2[rel, h, c]
// (hidden2 [R,N,H] never materialized)
__global__ void edge2_kernel(const int* __restrict__ rel,
                             const int* __restrict__ fr,
                             const int* __restrict__ to,
                             const float* __restrict__ deg,
                             const float* __restrict__ hidden1,
                             const float* __restrict__ w2,
                             float* __restrict__ out) {
    long long gid = (long long)blockIdx.x * blockDim.x + threadIdx.x;
    if (gid >= (long long)EE * CC) return;
    int e = (int)(gid >> 4);
    int c = (int)(gid & 15);
    int r = rel[e], f = fr[e], t = to[e];
    float val = 1.0f / deg[(long long)r * NN + f];
    const float* hrow = hidden1 + (long long)t * HH;     // broadcast across 16 lanes
    const float* wcol = w2 + (long long)r * HH * CC + c; // coalesced across lanes
    float acc = 0.0f;
    #pragma unroll
    for (int h = 0; h < HH; ++h)
        acc += hrow[h] * wcol[h * CC];
    atomicAdd(&out[(long long)f * CC + c], val * acc);
}

// ---------------------------------------------------------------------------

extern "C" void kernel_launch(void* const* d_in, const int* in_sizes, int n_in,
                              void* d_out, int out_size, void* d_ws, size_t ws_size,
                              hipStream_t stream) {
    const float* embeddings = (const float*)d_in[0];
    const float* comps1     = (const float*)d_in[1];
    const float* bases1     = (const float*)d_in[2];
    const float* comps2     = (const float*)d_in[3];
    const float* bases2     = (const float*)d_in[4];
    const float* bias1      = (const float*)d_in[5];
    const float* bias2      = (const float*)d_in[6];
    const int*   rel        = (const int*)d_in[7];
    const int*   fr         = (const int*)d_in[8];
    const int*   to         = (const int*)d_in[9];
    float* out = (float*)d_out;

    // workspace carve-out (256B aligned)
    char*  base = (char*)d_ws;
    size_t off  = 0;
    auto alloc = [&](size_t bytes) {
        void* p = base + off;
        off = (off + bytes + 255) & ~(size_t)255;
        return p;
    };
    unsigned short* embB    = (unsigned short*)alloc((size_t)NN * EMB * 2);  // 25.6 MB
    unsigned short* w1T     = (unsigned short*)alloc((size_t)RR * HH * EMB * 2);
    float*          w2      = (float*)alloc((size_t)RR * HH * CC * 4);
    float*          deg     = (float*)alloc((size_t)RR * NN * 4);            // 12.8 MB
    float*          hidden1 = (float*)alloc((size_t)NN * HH * 4);            // 12.8 MB
    float*          xw      = (float*)alloc((size_t)RR * NN * HH * 4);       // 409.6 MB
    (void)ws_size; (void)n_in; (void)in_sizes; (void)out_size;

    const int T = 256;

    // 1) degree (edge normalization denominators)
    {
        long long n = (long long)RR * NN;
        zero_f32<<<(unsigned)((n + T - 1) / T), T, 0, stream>>>(deg, n);
        degree_kernel<<<(EE + T - 1) / T, T, 0, stream>>>(rel, fr, deg);
    }
    // 2) bf16 operand prep for the WMMA GEMM
    {
        long long n = (long long)NN * EMB;
        conv_bf16_kernel<<<(unsigned)((n + T - 1) / T), T, 0, stream>>>(embeddings, embB, n);
        build_w1t_kernel<<<RR, T, 0, stream>>>(comps1, bases1, w1T);
        build_w2_kernel<<<RR, T, 0, stream>>>(comps2, bases2, w2);
    }
    // 3) xw[r] = emb @ w1[r]  — v_wmma_f32_16x16x32_bf16, B pinned in VGPRs
    {
        dim3 grid(GEMM_BLOCKS_X, RR);
        gemm_xw_kernel<<<grid, 256, 0, stream>>>(embB, w1T, xw);
    }
    // 4) layer-1 scatter + activation
    {
        long long n = (long long)NN * HH;
        zero_f32<<<(unsigned)((n + T - 1) / T), T, 0, stream>>>(hidden1, n);
        long long work = (long long)EE * HH;
        edge1_kernel<<<(unsigned)((work + T - 1) / T), T, 0, stream>>>(rel, fr, to, deg, xw, hidden1);
        relu_bias_kernel<<<(unsigned)((n + T - 1) / T), T, 0, stream>>>(hidden1, bias1);
    }
    // 5) layer-2: per-edge fused contraction with w2 (no hidden2 buffer)
    {
        long long n = (long long)NN * CC;
        out_init_kernel<<<(unsigned)((n + T - 1) / T), T, 0, stream>>>(out, bias2);
        long long work = (long long)EE * CC;
        edge2_kernel<<<(unsigned)((work + T - 1) / T), T, 0, stream>>>(rel, fr, to, deg, hidden1, w2, out);
    }
}
